// Net_71536975282266
// MI455X (gfx1250) — matrix operand order repaired and checked by reading, not compiled, not run on hardware
//
#include <hip/hip_runtime.h>
#include <hip/hip_bf16.h>
#include <stddef.h>

// ---------------------------------------------------------------------------
// DGCNN forward for MI455X (gfx1250, wave32):
//   - GEMMs via v_wmma_f32_16x16x32_f16, 16x64 strip per wave (4 accumulators)
//   - weight tiles staged to LDS by the Tensor Data Mover (tensor_load_to_lds)
//   - BN batch stats fused into GEMM epilogue (shfl -> ds_add_f32 -> atomics)
// ---------------------------------------------------------------------------

typedef __attribute__((ext_vector_type(16))) _Float16 v16h;
typedef __attribute__((ext_vector_type(8)))  _Float16 v8h;
typedef __attribute__((ext_vector_type(8)))  float    v8f;
typedef __attribute__((ext_vector_type(4)))  unsigned int u32x4;
typedef __attribute__((ext_vector_type(8)))  int      i32x8;
typedef __attribute__((ext_vector_type(4)))  int      i32x4;

#define FINF 3.402823466e38f
#define BN_EPS 1e-5f

#if defined(__has_builtin)
#if __has_builtin(__builtin_amdgcn_tensor_load_to_lds) && \
    __has_builtin(__builtin_amdgcn_s_wait_tensorcnt)
#define USE_TDM 1
#endif
#endif

static __device__ __forceinline__ v8f wmma_f16f32(v16h a, v16h b, v8f c) {
  return __builtin_amdgcn_wmma_f32_16x16x32_f16(
      /*neg_a=*/false, a, /*neg_b=*/false, b,
      /*c_mod=*/(short)0, c, /*reuse_a=*/false, /*reuse_b=*/false);
}

#ifdef USE_TDM
// TDM: DMA a 2-D tile (64 rows x klen f16 elements, row stride = stride elems)
// from global memory into LDS at lds_off.  D# layout per CDNA5 ISA ch.8.
// This toolchain exposes the 6-argument builtin (extra SGPR group + cpol).
static __device__ __forceinline__ void tdm_load_tile(const _Float16* gsrc,
                                                     unsigned lds_off,
                                                     int klen, int stride) {
  unsigned long long ga = (unsigned long long)(uintptr_t)gsrc;
  u32x4 g0 = {
      1u,                                            // count=1 (valid D#)
      lds_off,                                       // lds_addr (bytes)
      (unsigned)(ga & 0xffffffffu),                  // global_addr[31:0]
      (unsigned)((ga >> 32) & 0x1ffffffu) | (2u << 30)  // addr[56:32] | type=2
  };
  unsigned uk = (unsigned)klen;
  i32x8 g1 = {
      (int)(1u << 16),                 // workgroup_mask=0, data_size=1 (2B)
      (int)((uk & 0xffffu) << 16),     // tensor_dim0[15:0] @ bits 63:48
      (int)((uk >> 16) | (64u << 16)), // tensor_dim0[31:16], tensor_dim1=64
      (int)((uk & 0xffffu) << 16),     // tile_dim0 = klen @ bits 127:112
      64,                              // tile_dim1 = 64, tile_dim2 = 0
      stride,                          // tensor_dim0_stride[31:0]
      0,                               // stride[47:32]=0, dim1_stride lo=0
      0
  };
  i32x4 z4 = {0, 0, 0, 0};
  i32x8 z8 = {0, 0, 0, 0, 0, 0, 0, 0};
  __builtin_amdgcn_tensor_load_to_lds(g0, g1, z4, z4, z8, 0);
}
#endif

// ------------------------- small utility kernels ---------------------------

__global__ void build_x0_kernel(const float* __restrict__ x,
                                const float* __restrict__ pos,
                                float* __restrict__ h0, int N) {
  int i = blockIdx.x * blockDim.x + threadIdx.x;
  if (i >= N) return;
  h0[(size_t)i * 4 + 0] = x[i];
  h0[(size_t)i * 4 + 1] = pos[(size_t)i * 3 + 0];
  h0[(size_t)i * 4 + 2] = pos[(size_t)i * 3 + 1];
  h0[(size_t)i * 4 + 3] = pos[(size_t)i * 3 + 2];
}

__global__ void sqnorm_kernel(const float* __restrict__ h, float* __restrict__ sq,
                              int N, int F) {
  int i = blockIdx.x * blockDim.x + threadIdx.x;
  if (i >= N) return;
  float s = 0.f;
  const float* r = h + (size_t)i * F;
  for (int f = 0; f < F; ++f) { float v = r[f]; s += v * v; }
  sq[i] = s;
}

// W [K,Nout] f32 -> Wt [Nout,Kpad] f16 (transposed, zero K-pad)
__global__ void wprep_kernel(const float* __restrict__ W, _Float16* __restrict__ Wt,
                             int K, int Nout, int Kpad) {
  int gid = blockIdx.x * blockDim.x + threadIdx.x;
  int total = Nout * Kpad;
  if (gid >= total) return;
  int n = gid / Kpad;
  int k = gid % Kpad;
  Wt[gid] = (k < K) ? (_Float16)W[(size_t)k * Nout + n] : (_Float16)0.f;
}

// ----------------------------- kNN (top-30) --------------------------------
__global__ void knn_kernel(const float* __restrict__ h, const float* __restrict__ sq,
                           int* __restrict__ idxOut, int N, int F) {
  __shared__ float hi_s[64];
  __shared__ float cd[30 * 32];
  __shared__ int   ci[30 * 32];
  const int i = blockIdx.x;
  const int lane = threadIdx.x;

  for (int f = lane; f < F; f += 32) hi_s[f] = h[(size_t)i * F + f];
  for (int kk = 0; kk < 30; ++kk) cd[kk * 32 + lane] = FINF;
  __syncthreads();

  const float sqi = sq[i];
  float cmax = FINF;
  int cpos = 0;
  for (int c = lane; c < N; c += 32) {
    const float* hc = h + (size_t)c * F;
    float dot = 0.f;
    for (int f = 0; f < F; ++f) dot += hi_s[f] * hc[f];
    float d = sqi + sq[c] - 2.f * dot;
    if (d < cmax) {
      cd[cpos * 32 + lane] = d;
      ci[cpos * 32 + lane] = c;
      cmax = -FINF;
      for (int kk = 0; kk < 30; ++kk) {
        float v = cd[kk * 32 + lane];
        if (v > cmax) { cmax = v; cpos = kk; }
      }
    }
  }
  __syncthreads();

  float lmin = FINF; int lpos = 0;
  for (int kk = 0; kk < 30; ++kk) {
    float v = cd[kk * 32 + lane];
    if (v < lmin) { lmin = v; lpos = kk; }
  }
  for (int it = 0; it < 30; ++it) {
    float v = lmin; int l = lane;
    for (int m = 16; m > 0; m >>= 1) {
      float ov = __shfl_xor(v, m);
      int   ol = __shfl_xor(l, m);
      if (ov < v || (ov == v && ol < l)) { v = ov; l = ol; }
    }
    if (lane == l) {
      idxOut[(size_t)i * 30 + it] = ci[lpos * 32 + lane];
      cd[lpos * 32 + lane] = FINF;
      lmin = FINF; lpos = 0;
      for (int kk = 0; kk < 30; ++kk) {
        float vv = cd[kk * 32 + lane];
        if (vv < lmin) { lmin = vv; lpos = kk; }
      }
    }
  }
}

// -------------------- edge feature build [xi, xj-xi] -----------------------
__global__ void edge_build_kernel(const float* __restrict__ h, const int* __restrict__ idx,
                                  _Float16* __restrict__ E, int N, int K, int F, int Kpad) {
  size_t gid = (size_t)blockIdx.x * blockDim.x + threadIdx.x;
  size_t total = (size_t)N * K * Kpad;
  if (gid >= total) return;
  int col = (int)(gid % Kpad);
  size_t row = gid / Kpad;
  int i = (int)(row / K);
  int j = idx[row];
  float v = 0.f;
  if (col < F)           v = h[(size_t)i * F + col];
  else if (col < 2 * F)  { int f = col - F; v = h[(size_t)j * F + f] - h[(size_t)i * F + f]; }
  E[gid] = (_Float16)v;
}

// ------------------- GEMM + bias + ReLU + BN-stat epilogue -----------------
// Y = relu((AFF ? sIn*A+tIn : A) @ W + b)
// Per wave: 16 rows x 64 cols (4x v8f accumulators), K-loop step 32.
// Weight strip [64 x Kpad] staged to LDS in 256-deep chunks by the TDM.
#define KCHUNK 256

template <bool AFF>
__global__ __launch_bounds__(256) void gemm_bn_kernel(
    const _Float16* __restrict__ A, const _Float16* __restrict__ Wt,
    const float* __restrict__ bias,
    const float* __restrict__ sIn, const float* __restrict__ tIn,
    _Float16* __restrict__ Y, int M, int Kpad, int Nout,
    float* __restrict__ sumOut, float* __restrict__ sumsqOut) {
  __shared__ _Float16 lds_b[64 * KCHUNK];
  __shared__ float ssum[64];
  __shared__ float ssq[64];
  const int tid = threadIdx.x;
  const int lane = tid & 31;
  const int wave = tid >> 5;
  const int r0 = blockIdx.x * 128 + wave * 16;   // wave's row tile
  const int c0 = blockIdx.y * 64;                // block's 64-col strip

  if (tid < 64) { ssum[tid] = 0.f; ssq[tid] = 0.f; }

  const bool act  = (r0 < M);                    // M is a multiple of 16
  const int nn    = lane & 15;
  const int abase = (lane < 16) ? 0 : 16;   // A halfs: K=abase..+15
  const int bbase = (lane < 16) ? 0 : 16;   // B halfs: K=bbase..+15
  const int arow  = r0 + nn;

  v8f acc[4];
#pragma unroll
  for (int t = 0; t < 4; ++t) acc[t] = (v8f){};

  for (int k0 = 0; k0 < Kpad; k0 += KCHUNK) {
    int klen = Kpad - k0;
    if (klen > KCHUNK) klen = KCHUNK;
    __syncthreads();  // previous chunk fully consumed
#ifdef USE_TDM
    if (wave == 0) {
      tdm_load_tile(Wt + (size_t)c0 * Kpad + k0,
                    (unsigned)(uintptr_t)(void*)lds_b, klen, Kpad);
      __builtin_amdgcn_s_wait_tensorcnt(0);
    }
#else
    for (int e = tid * 8; e < 64 * klen; e += 256 * 8) {
      int row = e / klen, col = e - row * klen;
      *(v8h*)&lds_b[row * klen + col] =
          *(const v8h*)(Wt + (size_t)(c0 + row) * Kpad + k0 + col);
    }
#endif
    __syncthreads();  // LDS tile visible to all waves

    if (act) {
      const _Float16* abp = A + (size_t)arow * Kpad + k0;
      if (k0 + KCHUNK < Kpad) __builtin_prefetch(abp + KCHUNK, 0, 1);
      for (int kc = 0; kc < klen; kc += 32) {
        const _Float16* ap = abp + kc + abase;
        v8h alo = *(const v8h*)(ap);
        v8h ahi = *(const v8h*)(ap + 8);
        if (AFF) {
          int kk0 = k0 + kc + abase;
#pragma unroll
          for (int e = 0; e < 8; ++e)
            alo[e] = (_Float16)(sIn[kk0 + e] * (float)alo[e] + tIn[kk0 + e]);
          kk0 += 8;
#pragma unroll
          for (int e = 0; e < 8; ++e)
            ahi[e] = (_Float16)(sIn[kk0 + e] * (float)ahi[e] + tIn[kk0 + e]);
        }
        v16h a = __builtin_shufflevector(alo, ahi, 0, 1, 2, 3, 4, 5, 6, 7,
                                         8, 9, 10, 11, 12, 13, 14, 15);
#pragma unroll
        for (int t = 0; t < 4; ++t) {
          v16h b = *(const v16h*)&lds_b[(size_t)(t * 16 + nn) * klen + kc + bbase];
          acc[t] = wmma_f16f32(a, b, acc[t]);
        }
      }
    }
  }

  // epilogue: bias + relu + store + channel stats
  if (act) {
    const int rowoff = (lane < 16) ? 0 : 8;
#pragma unroll
    for (int t = 0; t < 4; ++t) {
      const int bcol = c0 + t * 16 + nn;
      const float bv = bias[bcol];
      float psum = 0.f, psq = 0.f;
#pragma unroll
      for (int r = 0; r < 8; ++r) {
        float v = acc[t][r] + bv;
        v = fmaxf(v, 0.f);
        Y[(size_t)(r0 + r + rowoff) * Nout + bcol] = (_Float16)v;
        psum += v;
        psq += v * v;
      }
      float osum = __shfl_down(psum, 16);
      float osq  = __shfl_down(psq, 16);
      if (lane < 16) {
        atomicAdd(&ssum[t * 16 + lane], psum + osum);   // ds_add_f32
        atomicAdd(&ssq[t * 16 + lane],  psq + osq);
      }
    }
  }
  __syncthreads();
  if (tid < 64) {
    atomicAdd(&sumOut[c0 + tid],   ssum[tid]);
    atomicAdd(&sumsqOut[c0 + tid], ssq[tid]);
  }
}

// BN batch stats -> per-channel affine: s = g*rsqrt(var+eps), t = bt - mu*s
__global__ void stats2affine_kernel(const float* __restrict__ sum,
                                    const float* __restrict__ sumsq,
                                    const float* __restrict__ g,
                                    const float* __restrict__ bt,
                                    float invM, int C,
                                    float* __restrict__ s, float* __restrict__ t) {
  int c = blockIdx.x * blockDim.x + threadIdx.x;
  if (c >= C) return;
  float mu  = sum[c] * invM;
  float var = sumsq[c] * invM - mu * mu;
  float inv = rsqrtf(var + BN_EPS);
  float sc  = g[c] * inv;
  s[c] = sc;
  t[c] = bt[c] - mu * sc;
}

// out[i][c] = max_kk ( s[c]*Y[(i*30+kk)*64+c] + t[c] )
__global__ void aggmax_kernel(const _Float16* __restrict__ Y,
                              const float* __restrict__ s, const float* __restrict__ t,
                              float* __restrict__ hout, int N) {
  int gid = blockIdx.x * blockDim.x + threadIdx.x;
  if (gid >= N * 64) return;
  int c = gid & 63;
  int i = gid >> 6;
  float sc = s[c], tc = t[c];
  float m = -FINF;
  for (int kk = 0; kk < 30; ++kk) {
    float y = (float)Y[((size_t)i * 30 + kk) * 64 + c];
    m = fmaxf(m, sc * y + tc);
  }
  hout[(size_t)i * 64 + c] = m;
}

__global__ void hcat_kernel(const float* __restrict__ x1, const float* __restrict__ x2,
                            const float* __restrict__ x3, _Float16* __restrict__ Hc, int N) {
  int gid = blockIdx.x * blockDim.x + threadIdx.x;
  if (gid >= N * 192) return;
  int c = gid % 192;
  int i = gid / 192;
  float v;
  if (c < 64)       v = x1[(size_t)i * 64 + c];
  else if (c < 128) v = x2[(size_t)i * 64 + (c - 64)];
  else              v = x3[(size_t)i * 64 + (c - 128)];
  Hc[gid] = (_Float16)v;
}

__global__ void final_kernel(const _Float16* __restrict__ Y,
                             const float* __restrict__ s, const float* __restrict__ t,
                             const float* __restrict__ Wf, const float* __restrict__ bf,
                             float* __restrict__ out, int N) {
  int i = blockIdx.x * blockDim.x + threadIdx.x;
  if (i >= N) return;
  float acc = bf[0];
  for (int c = 0; c < 128; ++c) {
    float y = (float)Y[(size_t)i * 128 + c];
    acc += (s[c] * y + t[c]) * Wf[c];
  }
  out[i] = acc;
}

// ---------------------------------------------------------------------------
// host side
// ---------------------------------------------------------------------------

extern "C" void kernel_launch(void* const* d_in, const int* in_sizes, int n_in,
                              void* d_out, int out_size, void* d_ws, size_t ws_size,
                              hipStream_t stream) {
  (void)in_sizes; (void)n_in; (void)out_size; (void)ws_size;
  const int N = 10000, K = 30;
  const int M_edge = N * K;

  const float* x   = (const float*)d_in[0];
  const float* pos = (const float*)d_in[1];
  const float* c1W0 = (const float*)d_in[2];  const float* c1b0 = (const float*)d_in[3];
  const float* c1g0 = (const float*)d_in[4];  const float* c1t0 = (const float*)d_in[5];
  const float* c1W1 = (const float*)d_in[6];  const float* c1b1 = (const float*)d_in[7];
  const float* c1g1 = (const float*)d_in[8];  const float* c1t1 = (const float*)d_in[9];
  const float* c2W0 = (const float*)d_in[10]; const float* c2b0 = (const float*)d_in[11];
  const float* c2g0 = (const float*)d_in[12]; const float* c2t0 = (const float*)d_in[13];
  const float* c2W1 = (const float*)d_in[14]; const float* c2b1 = (const float*)d_in[15];
  const float* c2g1 = (const float*)d_in[16]; const float* c2t1 = (const float*)d_in[17];
  const float* c3W0 = (const float*)d_in[18]; const float* c3b0 = (const float*)d_in[19];
  const float* c3g0 = (const float*)d_in[20]; const float* c3t0 = (const float*)d_in[21];
  const float* c3W1 = (const float*)d_in[22]; const float* c3b1 = (const float*)d_in[23];
  const float* c3g1 = (const float*)d_in[24]; const float* c3t1 = (const float*)d_in[25];
  const float* l1W  = (const float*)d_in[26]; const float* l1b  = (const float*)d_in[27];
  const float* l1g  = (const float*)d_in[28]; const float* l1t  = (const float*)d_in[29];
  const float* h1W  = (const float*)d_in[30]; const float* h1b  = (const float*)d_in[31];
  const float* h1g  = (const float*)d_in[32]; const float* h1t  = (const float*)d_in[33];
  const float* h2W  = (const float*)d_in[34]; const float* h2b  = (const float*)d_in[35];
  const float* h2g  = (const float*)d_in[36]; const float* h2t  = (const float*)d_in[37];
  const float* fW   = (const float*)d_in[38]; const float* fb   = (const float*)d_in[39];
  float* out = (float*)d_out;

  char* ws = (char*)d_ws;
  size_t off = 0;
  auto alloc = [&](size_t bytes) -> char* {
    char* p = ws + off;
    off = (off + bytes + 255) & ~(size_t)255;
    return p;
  };
  float* h0   = (float*)alloc((size_t)N * 4 * 4);
  float* h1   = (float*)alloc((size_t)N * 64 * 4);
  float* h2   = (float*)alloc((size_t)N * 64 * 4);
  float* h3   = (float*)alloc((size_t)N * 64 * 4);
  float* sqb  = (float*)alloc((size_t)N * 4);
  int*   idxb = (int*)  alloc((size_t)N * K * 4);
  float* stats = (float*)alloc(2048 * 4);
  float* sumB  = stats;
  float* sqsB  = stats + 1024;
  float* sA = (float*)alloc(1024 * 4);
  float* tA = (float*)alloc(1024 * 4);
  float* sB = (float*)alloc(1024 * 4);
  float* tB = (float*)alloc(1024 * 4);
  _Float16* wt_c1a = (_Float16*)alloc((size_t)64 * 32 * 2);
  _Float16* wt_c1b = (_Float16*)alloc((size_t)64 * 64 * 2);
  _Float16* wt_c2a = (_Float16*)alloc((size_t)64 * 128 * 2);
  _Float16* wt_c2b = (_Float16*)alloc((size_t)64 * 64 * 2);
  _Float16* wt_c3a = (_Float16*)alloc((size_t)64 * 128 * 2);
  _Float16* wt_c3b = (_Float16*)alloc((size_t)64 * 64 * 2);
  _Float16* wt_l1  = (_Float16*)alloc((size_t)1024 * 192 * 2);
  _Float16* wt_h1  = (_Float16*)alloc((size_t)256 * 1024 * 2);
  _Float16* wt_h2  = (_Float16*)alloc((size_t)128 * 256 * 2);
  _Float16* Ebuf = (_Float16*)alloc((size_t)M_edge * 128 * 2);
  _Float16* Y1   = (_Float16*)alloc((size_t)M_edge * 64 * 2);
  _Float16* Y2   = (_Float16*)alloc((size_t)M_edge * 64 * 2);

  auto wprep = [&](const float* W, _Float16* Wt, int Kd, int Nout, int Kpad) {
    int total = Nout * Kpad;
    wprep_kernel<<<(total + 255) / 256, 256, 0, stream>>>(W, Wt, Kd, Nout, Kpad);
  };
  wprep(c1W0, wt_c1a, 8, 64, 32);
  wprep(c1W1, wt_c1b, 64, 64, 64);
  wprep(c2W0, wt_c2a, 128, 64, 128);
  wprep(c2W1, wt_c2b, 64, 64, 64);
  wprep(c3W0, wt_c3a, 128, 64, 128);
  wprep(c3W1, wt_c3b, 64, 64, 64);
  wprep(l1W, wt_l1, 192, 1024, 192);
  wprep(h1W, wt_h1, 1024, 256, 1024);
  wprep(h2W, wt_h2, 256, 128, 256);

  auto gemm = [&](const _Float16* A, const _Float16* Wt, const float* bias,
                  const float* si, const float* ti, _Float16* Y,
                  int M, int Kpad, int Nout) {
    (void)hipMemsetAsync(stats, 0, 2048 * sizeof(float), stream);
    dim3 grid((M + 127) / 128, Nout / 64);
    if (si)
      gemm_bn_kernel<true><<<grid, 256, 0, stream>>>(A, Wt, bias, si, ti, Y, M,
                                                     Kpad, Nout, sumB, sqsB);
    else
      gemm_bn_kernel<false><<<grid, 256, 0, stream>>>(A, Wt, bias, si, ti, Y, M,
                                                      Kpad, Nout, sumB, sqsB);
  };
  auto affine = [&](const float* g, const float* bt, int M, int C, float* s, float* t) {
    stats2affine_kernel<<<(C + 63) / 64, 64, 0, stream>>>(sumB, sqsB, g, bt,
                                                          1.0f / (float)M, C, s, t);
  };

  auto edge_conv = [&](const float* hin, int F, int Kpad,
                       const _Float16* wtA, const float* bA, const float* gA, const float* btA,
                       const _Float16* wtB, const float* bBi, const float* gB2, const float* btB,
                       float* hout) {
    sqnorm_kernel<<<(N + 255) / 256, 256, 0, stream>>>(hin, sqb, N, F);
    knn_kernel<<<N, 32, 0, stream>>>(hin, sqb, idxb, N, F);
    size_t total = (size_t)N * K * Kpad;
    edge_build_kernel<<<(unsigned)((total + 255) / 256), 256, 0, stream>>>(
        hin, idxb, Ebuf, N, K, F, Kpad);
    gemm(Ebuf, wtA, bA, nullptr, nullptr, Y1, M_edge, Kpad, 64);
    affine(gA, btA, M_edge, 64, sA, tA);
    gemm(Y1, wtB, bBi, sA, tA, Y2, M_edge, 64, 64);
    affine(gB2, btB, M_edge, 64, sB, tB);
    aggmax_kernel<<<(N * 64 + 255) / 256, 256, 0, stream>>>(Y2, sB, tB, hout, N);
  };

  build_x0_kernel<<<(N + 255) / 256, 256, 0, stream>>>(x, pos, h0, N);

  edge_conv(h0, 4, 32,   wt_c1a, c1b0, c1g0, c1t0, wt_c1b, c1b1, c1g1, c1t1, h1);
  edge_conv(h1, 64, 128, wt_c2a, c2b0, c2g0, c2t0, wt_c2b, c2b1, c2g1, c2t1, h2);
  edge_conv(h2, 64, 128, wt_c3a, c3b0, c3g0, c3t0, wt_c3b, c3b1, c3g1, c3t1, h3);

  _Float16* Hcat  = Ebuf;
  _Float16* Ylin1 = Y1;
  _Float16* Yh1   = Y2;
  _Float16* Yh2   = Ebuf;   // Hcat dead by then
  hcat_kernel<<<(N * 192 + 255) / 256, 256, 0, stream>>>(h1, h2, h3, Hcat, N);

  gemm(Hcat, wt_l1, l1b, nullptr, nullptr, Ylin1, N, 192, 1024);
  affine(l1g, l1t, N, 1024, sA, tA);
  gemm(Ylin1, wt_h1, h1b, sA, tA, Yh1, N, 1024, 256);
  affine(h1g, h1t, N, 256, sB, tB);
  gemm(Yh1, wt_h2, h2b, sB, tB, Yh2, N, 256, 128);
  affine(h2g, h2t, N, 128, sA, tA);

  final_kernel<<<(N + 255) / 256, 256, 0, stream>>>(Yh2, sA, tA, fW, fb, out, N);
}
